// CP_ClDiceLoss_53094385713447
// MI455X (gfx1250) — compile-verified, matrix operand order checked by reading, not run on hardware
//
#include <hip/hip_runtime.h>
#include <hip/hip_bf16.h>

typedef __attribute__((ext_vector_type(16))) _Float16 v16h;
typedef __attribute__((ext_vector_type(8)))  _Float16 v8h;
typedef __attribute__((ext_vector_type(8)))  float    v8f;

#define BATCH 8
#define HW    1024
#define IMG   (1024 * 1024)      // pixels per image
#define TOT   (BATCH * IMG)      // 8M

union HF { v16h v; v8h h[2]; };

__device__ inline v8f wmma_f16(v16h a, v16h b, v8f c) {
    return __builtin_amdgcn_wmma_f32_16x16x32_f16(false, a, false, b, (short)0, c, false, false);
}

// ---------------------------------------------------------------------------
// init m1/m2 (min-distance arrays) to +inf bit pattern
// ---------------------------------------------------------------------------
__global__ __launch_bounds__(256) void init_min_k(unsigned int* mm) {
    int i = blockIdx.x * 256 + threadIdx.x;     // 16384 entries
    if (i < 16384) mm[i] = 0x7f800000u;
}

// ---------------------------------------------------------------------------
// Phase A: p = sigmoid(pred1-pred0), t = onehot, f16 copies, row sums.
// grid = (1024 rows, 8 batches), block = 256 (4 px/thread)
// ---------------------------------------------------------------------------
__global__ __launch_bounds__(256) void phaseA_k(const float* __restrict__ pred,
                                                const int* __restrict__ target,
                                                float* __restrict__ p32, float* __restrict__ t32,
                                                _Float16* __restrict__ p16, _Float16* __restrict__ t16,
                                                float* __restrict__ a2, float* __restrict__ b2,
                                                float* __restrict__ row_pt, float* __restrict__ row_lg) {
    __shared__ float s0[256], s1[256], s2[256], s3[256];
    const int y = blockIdx.x, b = blockIdx.y, tid = threadIdx.x;
    const size_t rowbase = ((size_t)b * HW + y) * HW;
    const float* pr0 = pred + ((size_t)(b * 2 + 0) * HW + y) * HW;
    const float* pr1 = pred + ((size_t)(b * 2 + 1) * HW + y) * HW;
    const int* tg = target + rowbase;

    float sp2 = 0.f, st = 0.f, sptv = 0.f, slg = 0.f;
    #pragma unroll
    for (int k = 0; k < 4; ++k) {
        int x = tid * 4 + k;
        float z = pr1[x] - pr0[x];
        float p = 1.0f / (1.0f + __expf(-z));
        int   g = tg[x];
        int   gc = g < 0 ? 0 : (g > 1 ? 1 : g);
        float t = (gc == 1) ? 1.0f : 0.0f;
        float valid = (g != 255) ? 1.0f : 0.0f;
        sp2  += p * p;
        st   += t;
        sptv += p * t * valid;
        slg  += (1.0f - t) * __logf(1.0f - p + 1e-6f);
        p32[rowbase + x] = p;
        t32[rowbase + x] = t;
        p16[rowbase + x] = (_Float16)p;
        t16[rowbase + x] = (_Float16)t;
    }
    s0[tid] = sp2; s1[tid] = st; s2[tid] = sptv; s3[tid] = slg;
    __syncthreads();
    for (int s = 128; s > 0; s >>= 1) {
        if (tid < s) { s0[tid] += s0[tid+s]; s1[tid] += s1[tid+s];
                       s2[tid] += s2[tid+s]; s3[tid] += s3[tid+s]; }
        __syncthreads();
    }
    if (tid == 0) {
        int r = b * HW + y;
        a2[r] = s0[0];      // sum p^2 over row
        b2[r] = s1[0];      // sum t   over row (t^2 == t)
        row_pt[r] = s2[0];
        row_lg[r] = s3[0];
    }
}

// ---------------------------------------------------------------------------
// Phase B: Hausdorff distance matrix via WMMA f16 GEMM (NT), fused min-reduce
// grid = (N/128, M/64, batch), block = 256 = 8 waves, wave tile 32x32
// ---------------------------------------------------------------------------
__global__ __launch_bounds__(256) void hausdorff_k(const _Float16* __restrict__ P,
                                                   const _Float16* __restrict__ T,
                                                   const float* __restrict__ a2,
                                                   const float* __restrict__ b2,
                                                   unsigned int* __restrict__ m1,
                                                   unsigned int* __restrict__ m2) {
    const int b   = blockIdx.z;
    const int tid = threadIdx.x;
    const int lane = tid & 31;
    const int w  = tid >> 5;
    const int wm = w >> 2, wn = w & 3;
    const int I0 = blockIdx.y * 64 + wm * 32;
    const int J0 = blockIdx.x * 128 + wn * 32;
    const int lh = lane & 15, hh = lane >> 4;

    const _Float16* Pb = P + ((size_t)b << 20);
    const _Float16* Tb = T + ((size_t)b << 20);
    const float* a2b = a2 + b * HW;
    const float* b2b = b2 + b * HW;
    unsigned int* m1b = m1 + b * HW;
    unsigned int* m2b = m2 + b * HW;

    // per-lane fragment base pointers (16-bit A layout: lanes0-15 K 0-7,16-23;
    // lanes16-31 K 8-15,24-31). Both operands row-major (NT GEMM).
    const _Float16* pa0 = Pb + (size_t)(I0 + lh) * HW + hh * 8;
    const _Float16* pa1 = pa0 + (size_t)16 * HW;
    const _Float16* pb0 = Tb + (size_t)(J0 + lh) * HW + hh * 8;
    const _Float16* pb1 = pb0 + (size_t)16 * HW;

    v8f c00 = {}, c01 = {}, c10 = {}, c11 = {};
    #pragma unroll 2
    for (int k = 0; k < HW; k += 32) {
        HF a0, a1, bb0, bb1;
        a0.h[0]  = *(const v8h*)(pa0 + k);  a0.h[1]  = *(const v8h*)(pa0 + k + 16);
        a1.h[0]  = *(const v8h*)(pa1 + k);  a1.h[1]  = *(const v8h*)(pa1 + k + 16);
        bb0.h[0] = *(const v8h*)(pb0 + k);  bb0.h[1] = *(const v8h*)(pb0 + k + 16);
        bb1.h[0] = *(const v8h*)(pb1 + k);  bb1.h[1] = *(const v8h*)(pb1 + k + 16);
        c00 = wmma_f16(a0.v, bb0.v, c00);
        c01 = wmma_f16(a0.v, bb1.v, c01);
        c10 = wmma_f16(a1.v, bb0.v, c10);
        c11 = wmma_f16(a1.v, bb1.v, c11);
    }

    auto emit = [&](v8f s, int Ib, int Jb) {
        int j = Jb + lh;
        float b2j = b2b[j];
        float dv[8];
        float colmin = 3.0e38f;
        #pragma unroll
        for (int r = 0; r < 8; ++r) {
            float a2i = a2b[Ib + hh * 8 + r];
            float dd = __fsqrt_rn(fmaxf(a2i + b2j - 2.0f * s[r], 0.0f));
            dv[r] = dd;
            colmin = fminf(colmin, dd);
        }
        atomicMin(&m1b[j], __float_as_uint(colmin));   // min over rows i
        #pragma unroll
        for (int r = 0; r < 8; ++r) {
            float v = dv[r];
            v = fminf(v, __shfl_xor(v, 1));
            v = fminf(v, __shfl_xor(v, 2));
            v = fminf(v, __shfl_xor(v, 4));
            v = fminf(v, __shfl_xor(v, 8));
            if (lh == 0) atomicMin(&m2b[Ib + hh * 8 + r], __float_as_uint(v)); // min over cols j
        }
    };
    emit(c00, I0, J0);
    emit(c01, I0, J0 + 16);
    emit(c10, I0 + 16, J0);
    emit(c11, I0 + 16, J0 + 16);
}

// ---------------------------------------------------------------------------
// stencils: erode = min over 5-pt cross (OOB ignored); dilate = 3x3 max
// grid = (1024 rows, 8 batches), block = 256, 4 px/thread
// ---------------------------------------------------------------------------
__global__ __launch_bounds__(256) void erode_k(const float* __restrict__ in, float* __restrict__ out) {
    int y = blockIdx.x, b = blockIdx.y;
    const float* img = in + ((size_t)b << 20);
    float* o = out + ((size_t)b << 20);
    size_t row = (size_t)y << 10;
    #pragma unroll
    for (int k = 0; k < 4; ++k) {
        int x = threadIdx.x * 4 + k;
        float c = img[row + x];
        float vmin = c;
        if (y > 0)    vmin = fminf(vmin, img[row - HW + x]);
        if (y < 1023) vmin = fminf(vmin, img[row + HW + x]);
        float hmin = c;
        if (x > 0)    hmin = fminf(hmin, img[row + x - 1]);
        if (x < 1023) hmin = fminf(hmin, img[row + x + 1]);
        o[row + x] = fminf(vmin, hmin);
    }
}

__global__ __launch_bounds__(256) void dilate_k(const float* __restrict__ in, float* __restrict__ out) {
    int y = blockIdx.x, b = blockIdx.y;
    const float* img = in + ((size_t)b << 20);
    float* o = out + ((size_t)b << 20);
    #pragma unroll
    for (int k = 0; k < 4; ++k) {
        int x = threadIdx.x * 4 + k;
        float m = -3.0e38f;
        #pragma unroll
        for (int dy = -1; dy <= 1; ++dy) {
            int yy = y + dy; if (yy < 0 || yy > 1023) continue;
            const float* rr = img + ((size_t)yy << 10);
            #pragma unroll
            for (int dx = -1; dx <= 1; ++dx) {
                int xx = x + dx; if (xx < 0 || xx > 1023) continue;
                m = fmaxf(m, rr[xx]);
            }
        }
        o[((size_t)y << 10) + x] = m;
    }
}

// skel = relu(src - open)
__global__ __launch_bounds__(256) void skel_init_k(const float* __restrict__ src,
                                                   const float* __restrict__ open_,
                                                   float* __restrict__ S) {
    size_t i0 = ((size_t)blockIdx.x * 256 + threadIdx.x) * 4;
    #pragma unroll
    for (int k = 0; k < 4; ++k) {
        size_t i = i0 + k;
        S[i] = fmaxf(src[i] - open_[i], 0.0f);
    }
}

// delta = relu(img - open); S += relu(delta - S*delta)
__global__ __launch_bounds__(256) void skel_update_k(const float* __restrict__ img,
                                                     const float* __restrict__ open_,
                                                     float* __restrict__ S) {
    size_t i0 = ((size_t)blockIdx.x * 256 + threadIdx.x) * 4;
    #pragma unroll
    for (int k = 0; k < 4; ++k) {
        size_t i = i0 + k;
        float d = fmaxf(img[i] - open_[i], 0.0f);
        float s = S[i];
        S[i] = s + fmaxf(d - s * d, 0.0f);
    }
}

// partial sums of S and S*other, deterministic per-block tree; grid = 4096
__global__ __launch_bounds__(256) void reduce2_k(const float* __restrict__ S,
                                                 const float* __restrict__ other,
                                                 float* __restrict__ out_s,
                                                 float* __restrict__ out_so) {
    __shared__ float r0[256], r1[256];
    int tid = threadIdx.x;
    size_t base = (size_t)blockIdx.x * 2048;
    float ss = 0.f, sso = 0.f;
    #pragma unroll
    for (int k = 0; k < 8; ++k) {
        size_t i = base + tid + (size_t)k * 256;
        float v = S[i];
        ss  += v;
        sso += v * other[i];
    }
    r0[tid] = ss; r1[tid] = sso;
    __syncthreads();
    for (int s = 128; s > 0; s >>= 1) {
        if (tid < s) { r0[tid] += r0[tid+s]; r1[tid] += r1[tid+s]; }
        __syncthreads();
    }
    if (tid == 0) { out_s[blockIdx.x] = r0[0]; out_so[blockIdx.x] = r1[0]; }
}

// ---------------------------------------------------------------------------
// final combine: one block, deterministic reductions over small arrays
// ---------------------------------------------------------------------------
__global__ __launch_bounds__(256) void final_k(const float* __restrict__ a2,
                                               const float* __restrict__ b2,
                                               const unsigned int* __restrict__ m1, // m1,m2 contiguous (16384)
                                               const float* __restrict__ row_pt,
                                               const float* __restrict__ row_lg,
                                               const float* __restrict__ pred_s,
                                               const float* __restrict__ pred_st,
                                               const float* __restrict__ true_s,
                                               const float* __restrict__ true_sp,
                                               float* __restrict__ out) {
    __shared__ float red[256];
    const int tid = threadIdx.x;

    auto bsum = [&](float v) -> float {
        red[tid] = v; __syncthreads();
        for (int s = 128; s > 0; s >>= 1) { if (tid < s) red[tid] += red[tid+s]; __syncthreads(); }
        float r = red[0]; __syncthreads(); return r;
    };
    auto bmax = [&](float v) -> float {
        red[tid] = v; __syncthreads();
        for (int s = 128; s > 0; s >>= 1) { if (tid < s) red[tid] = fmaxf(red[tid], red[tid+s]); __syncthreads(); }
        float r = red[0]; __syncthreads(); return r;
    };

    // boundary dice (per-batch)
    float bdsum = 0.f;
    for (int b = 0; b < BATCH; ++b) {
        float lpt = 0.f, lden = 0.f;
        for (int i = tid; i < HW; i += 256) {
            lpt  += row_pt[b * HW + i];
            lden += a2[b * HW + i] + b2[b * HW + i];
        }
        float pt  = bsum(lpt);
        float den = bsum(lden);
        bdsum += 1.0f - (2.0f * pt + 1.0f) / (den + 1.0f);
    }
    float bd = bdsum / (float)BATCH;

    // log term
    float llg = 0.f;
    for (int i = tid; i < BATCH * HW; i += 256) llg += row_lg[i];
    float lg = -bsum(llg) / (float)TOT;

    // hausdorff: max over m1 & m2 (contiguous)
    float lmx = 0.f;
    for (int i = tid; i < 16384; i += 256) lmx = fmaxf(lmx, __uint_as_float(m1[i]));
    float dh = bmax(lmx);

    // skeleton sums
    float a = 0.f, bb = 0.f, cc = 0.f, dd = 0.f;
    for (int i = tid; i < 4096; i += 256) {
        a  += pred_s[i]; bb += pred_st[i];
        cc += true_s[i]; dd += true_sp[i];
    }
    float Sp  = bsum(a);
    float Spt = bsum(bb);
    float St  = bsum(cc);
    float Stp = bsum(dd);

    float tprec = (Spt + 1.0f) / (Sp + 1.0f);
    float tsens = (Stp + 1.0f) / (St + 1.0f);
    float cl = 1.0f - 2.0f * (tprec * tsens) / (tprec + tsens);

    if (tid == 0) out[0] = 0.5f * (bd + dh + lg) + 0.5f * cl;
}

// ---------------------------------------------------------------------------
extern "C" void kernel_launch(void* const* d_in, const int* in_sizes, int n_in,
                              void* d_out, int out_size, void* d_ws, size_t ws_size,
                              hipStream_t stream) {
    const float* pred  = (const float*)d_in[0];
    const int*   targ  = (const int*)d_in[1];
    float* out = (float*)d_out;

    char* ws = (char*)d_ws;
    // small header region
    float* a2      = (float*)ws;            // 8192
    float* b2      = a2 + 8192;             // 8192
    unsigned int* m1 = (unsigned int*)(b2 + 8192); // 8192
    unsigned int* m2 = m1 + 8192;                  // 8192 (contiguous with m1)
    float* row_pt  = (float*)(m2 + 8192);   // 8192
    float* row_lg  = row_pt + 8192;         // 8192
    float* pred_s  = row_lg + 8192;         // 4096
    float* pred_st = pred_s + 4096;         // 4096
    float* true_s  = pred_st + 4096;        // 4096
    float* true_sp = true_s + 4096;         // 4096

    const size_t MB = 1ull << 20;
    float*    p32  = (float*)(ws + 1 * MB);            // 32 MB
    float*    t32  = (float*)(ws + 33 * MB);           // 32 MB
    _Float16* p16  = (_Float16*)(ws + 65 * MB);        // 16 MB
    _Float16* t16  = (_Float16*)(ws + 81 * MB);        // 16 MB
    // skel scratch; bufA aliases p16/t16 (GEMM completes before skel phase)
    float* bufA = (float*)(ws + 65 * MB);              // 32 MB
    float* bufB = (float*)(ws + 97 * MB);              // 32 MB
    float* bufC = (float*)(ws + 129 * MB);             // 32 MB
    float* bufS = (float*)(ws + 161 * MB);             // 32 MB

    dim3 blk(256);
    dim3 grid_row(HW, BATCH);
    dim3 grid_gemm(HW / 128, HW / 64, BATCH);
    int  grid_ew  = TOT / 1024;   // 8192 (4 px/thread)

    // Phase init + A
    init_min_k<<<64, blk, 0, stream>>>(m1);   // covers m1+m2 (16384 contiguous)
    phaseA_k<<<grid_row, blk, 0, stream>>>(pred, targ, p32, t32, p16, t16,
                                           a2, b2, row_pt, row_lg);

    // Phase B: WMMA hausdorff
    hausdorff_k<<<grid_gemm, blk, 0, stream>>>(p16, t16, a2, b2, m1, m2);

    // Phase C: soft_skel for both images
    auto soft_skel = [&](const float* src) {
        // E1 = erode(src) -> A ; open -> B ; S = relu(src - B)
        erode_k<<<grid_row, blk, 0, stream>>>(src, bufA);
        dilate_k<<<grid_row, blk, 0, stream>>>(bufA, bufB);
        skel_init_k<<<grid_ew, blk, 0, stream>>>(src, bufB, bufS);
        // iter1: img=A ; erode(A)->C ; dilate(C)->B ; update(A,B)
        erode_k<<<grid_row, blk, 0, stream>>>(bufA, bufC);
        dilate_k<<<grid_row, blk, 0, stream>>>(bufC, bufB);
        skel_update_k<<<grid_ew, blk, 0, stream>>>(bufA, bufB, bufS);
        // iter2: img=C ; erode(C)->A ; dilate(A)->B ; update(C,B)
        erode_k<<<grid_row, blk, 0, stream>>>(bufC, bufA);
        dilate_k<<<grid_row, blk, 0, stream>>>(bufA, bufB);
        skel_update_k<<<grid_ew, blk, 0, stream>>>(bufC, bufB, bufS);
        // iter3: img=A ; erode(A)->C ; dilate(C)->B ; update(A,B)
        erode_k<<<grid_row, blk, 0, stream>>>(bufA, bufC);
        dilate_k<<<grid_row, blk, 0, stream>>>(bufC, bufB);
        skel_update_k<<<grid_ew, blk, 0, stream>>>(bufA, bufB, bufS);
    };

    soft_skel(p32);
    reduce2_k<<<4096, blk, 0, stream>>>(bufS, t32, pred_s, pred_st);
    soft_skel(t32);
    reduce2_k<<<4096, blk, 0, stream>>>(bufS, p32, true_s, true_sp);

    // final combine
    final_k<<<1, blk, 0, stream>>>(a2, b2, m1, row_pt, row_lg,
                                   pred_s, pred_st, true_s, true_sp, out);
}
